// MultiHeadAttention_30253749633260
// MI455X (gfx1250) — compile-verified, matrix operand order checked by reading, not run on hardware
//
#include <hip/hip_runtime.h>

#define BB 2
#define SS 2048
#define DD 1024
#define HH 16
#define DKV 64
#define NEGV (-1e9f)

typedef __bf16 bf16;
typedef __attribute__((ext_vector_type(16))) __bf16 v16bf;
typedef __attribute__((ext_vector_type(8)))  __bf16 v8bf;
typedef __attribute__((ext_vector_type(8)))  float   v8f;

static __device__ __forceinline__ v8f wmma_bf16(v16bf a, v16bf b, v8f c) {
  // D = A(16x32 bf16) x B(32x16 bf16) + C(16x16 f32)
  return __builtin_amdgcn_wmma_f32_16x16x32_bf16(false, a, false, b, (short)0, c, false, false);
}

// Build an A/B fragment (16 bf16 per lane) from a row-major [M][K] bf16 source.
// Caller passes p = row_base + cb, where cb = (lane&16)?8:0.
// Lane layout: elems 0..7 -> k=cb..cb+7, elems 8..15 -> k=cb+16..cb+23.
static __device__ __forceinline__ v16bf frag_bf16(const bf16* p) {
  v8bf lo = *(const v8bf*)p;
  v8bf hi = *(const v8bf*)(p + 16);
  v16bf r;
#pragma unroll
  for (int i = 0; i < 8; ++i) { r[i] = lo[i]; r[8 + i] = hi[i]; }
  return r;
}

// Same fragment, converting from an f32 source (global or LDS).
static __device__ __forceinline__ v16bf frag_f32(const float* p) {
  v16bf r;
#pragma unroll
  for (int i = 0; i < 8; ++i) { r[i] = (bf16)p[i]; r[8 + i] = (bf16)p[16 + i]; }
  return r;
}

static __device__ __forceinline__ float wave_max(float v) {
#pragma unroll
  for (int m = 16; m >= 1; m >>= 1) v = fmaxf(v, __shfl_xor(v, m, 32));
  return v;
}
static __device__ __forceinline__ float wave_sum(float v) {
#pragma unroll
  for (int m = 16; m >= 1; m >>= 1) v += __shfl_xor(v, m, 32);
  return v;
}

// ---------------- Kernel 0: weight transpose + f32->bf16 ([K][N] -> [N][K]) --
__global__ void __launch_bounds__(256) wt_kernel(
    const float* __restrict__ Wq, const float* __restrict__ Wk,
    const float* __restrict__ Wv, const float* __restrict__ Wo,
    bf16* __restrict__ wT) {
  const float* W = (blockIdx.y == 0) ? Wq : (blockIdx.y == 1) ? Wk
                  : (blockIdx.y == 2) ? Wv : Wo;
  bf16* dst = wT + (size_t)blockIdx.y * DD * DD;
  int idx = blockIdx.x * 256 + threadIdx.x;   // 0 .. 1048575
  int k = idx >> 10, n = idx & 1023;
  dst[(size_t)n * DD + k] = (bf16)W[(size_t)k * DD + n];
}

// ---------------- Kernel 1: QKV projections (X @ W + b), bf16 outputs --------
__global__ void __launch_bounds__(256) proj_kernel(
    const float* __restrict__ Xq, const float* __restrict__ Xk, const float* __restrict__ Xv,
    const bf16* __restrict__ wT,
    const float* __restrict__ bq, const float* __restrict__ bk, const float* __restrict__ bv,
    bf16* __restrict__ qbuf, bf16* __restrict__ kbuf, bf16* __restrict__ vT) {
  const int z = blockIdx.y;                  // 0=q,1=k,2=v
  const float* X   = (z == 0) ? Xq : (z == 1) ? Xk : Xv;
  const float* bia = (z == 0) ? bq : (z == 1) ? bk : bv;
  const bf16* Wt = wT + (size_t)z * DD * DD;

  int tid = threadIdx.x, wid = tid >> 5, lane = tid & 31;
  int gwid = blockIdx.x * 8 + wid;           // 0..4095 wave-tiles
  int m0 = (gwid >> 5) * 32;                 // 128 row tiles
  int n0 = (gwid & 31) * 32;                 // 32 col tiles
  int cb  = (lane & 16) ? 8 : 0;
  int l15 = lane & 15;
  v8f zz = {};
  v8f c00 = zz, c01 = zz, c10 = zz, c11 = zz;

  const float* aP0 = X  + (size_t)(m0 + l15)      * DD + cb;
  const float* aP1 = X  + (size_t)(m0 + 16 + l15) * DD + cb;
  const bf16*  bP0 = Wt + (size_t)(n0 + l15)      * DD + cb;
  const bf16*  bP1 = Wt + (size_t)(n0 + 16 + l15) * DD + cb;

  for (int kk = 0; kk < DD; kk += 32) {
    __builtin_prefetch(aP0 + kk + 64, 0, 0);
    __builtin_prefetch(aP1 + kk + 64, 0, 0);
    v16bf a0 = frag_f32(aP0 + kk);
    v16bf a1 = frag_f32(aP1 + kk);
    v16bf b0 = frag_bf16(bP0 + kk);
    v16bf b1 = frag_bf16(bP1 + kk);
    c00 = wmma_bf16(a0, b0, c00);
    c01 = wmma_bf16(a0, b1, c01);
    c10 = wmma_bf16(a1, b0, c10);
    c11 = wmma_bf16(a1, b1, c11);
  }
  int hi = (lane & 16) ? 8 : 0;
#pragma unroll
  for (int sj = 0; sj < 2; ++sj) {
    int ncol = n0 + sj * 16 + l15;
    float bval = bia[ncol];
#pragma unroll
    for (int si = 0; si < 2; ++si) {
      const v8f& cc = si ? (sj ? c11 : c10) : (sj ? c01 : c00);
#pragma unroll
      for (int r = 0; r < 8; ++r) {
        int mrow = m0 + si * 16 + r + hi;
        float v = cc[r] + bval;
        if (z == 0)      qbuf[(size_t)mrow * DD + ncol] = (bf16)v;
        else if (z == 1) kbuf[(size_t)mrow * DD + ncol] = (bf16)v;
        else {           // store V transposed: [ (b*H+h)*64 + d ][ s ]
          int b = mrow >> 11, s = mrow & 2047;
          int h = ncol >> 6,  d = ncol & 63;
          vT[((size_t)((b * HH + h) * DKV + d)) * SS + s] = (bf16)v;
        }
      }
    }
  }
}

// ---------------- Kernel 2: scores + softmax + attn output + P@V -------------
// One workgroup (8 waves) per (b, h, 16-query tile). 16x2048 f32 score strip
// lives entirely in LDS (128 KB of the WGP's 320 KB).
__global__ void __launch_bounds__(256) attn_kernel(
    const bf16* __restrict__ qbuf, const bf16* __restrict__ kbuf,
    const bf16* __restrict__ vT, const unsigned char* __restrict__ mask,
    bf16* __restrict__ ctxbuf, float* __restrict__ attn_out) {
  extern __shared__ char smem[];
  float* sc  = (float*)smem;                                  // [16][2048] f32
  bf16*  qt  = (bf16*)(smem + 16 * 2048 * 4);                 // [16][64] bf16
  float* ctx = (float*)(smem + 16 * 2048 * 4 + 16 * 64 * 2);  // [16][64] f32

  int tid = threadIdx.x, wid = tid >> 5, lane = tid & 31;
  int bh = blockIdx.x >> 7;            // 0..31
  int mt = blockIdx.x & 127;
  int b = bh >> 4, h = bh & 15;
  int m0 = mt * 16;

  { // stage q tile (already bf16)
    int i4 = tid * 4;
    int m = i4 >> 6, col = i4 & 63;
    const bf16* src = qbuf + (size_t)(b * SS + m0 + m) * DD + h * DKV + col;
    *(unsigned long long*)(qt + m * 64 + col) = *(const unsigned long long*)src;
  }
  __syncthreads();

  int cb = (lane & 16) ? 8 : 0;
  int l15 = lane & 15;
  int hi = (lane & 16) ? 8 : 0;

  v16bf aq0 = frag_bf16(qt + l15 * 64 + cb);        // q features 0..31
  v16bf aq1 = frag_bf16(qt + l15 * 64 + 32 + cb);   // q features 32..63

  // Phase 1: scores. Wave w covers keys [w*256, w*256+256).
  for (int nc = 0; nc < 16; ++nc) {
    int n0 = wid * 256 + nc * 16;
    const bf16* kb = kbuf + (size_t)(b * SS + n0 + l15) * DD + h * DKV + cb;
    __builtin_prefetch(kb + 16 * DD, 0, 0);
    v16bf bk0 = frag_bf16(kb);
    v16bf bk1 = frag_bf16(kb + 32);
    v8f c = {};
    c = wmma_bf16(aq0, bk0, c);
    c = wmma_bf16(aq1, bk1, c);
    int key = n0 + l15;
#pragma unroll
    for (int r = 0; r < 8; ++r) {
      int m = r + hi;
      float s = c[r] * 0.125f;                       // 1/sqrt(64)
      if (mask[(size_t)(b * SS + m0 + m) * SS + key]) s = NEGV;
      sc[m * 2048 + key] = s;
    }
  }
  __syncthreads();

  // Phase 2: softmax (2 rows per wave) + write attn output once.
#pragma unroll
  for (int e = 0; e < 2; ++e) {
    int r = wid * 2 + e;
    float* row = sc + r * 2048;
    float mx = -3.4e38f;
    for (int j = lane; j < 2048; j += 32) mx = fmaxf(mx, row[j]);
    mx = wave_max(mx);
    float sum = 0.f;
    for (int j = lane; j < 2048; j += 32) {
      float p = __expf(row[j] - mx);
      row[j] = p;
      sum += p;
    }
    sum = wave_sum(sum);
    float inv = 1.0f / sum;
    float* ao = attn_out + (size_t)(bh * SS + m0 + r) * SS;
    for (int j = lane; j < 2048; j += 32) {
      float p = row[j] * inv;
      row[j] = p;
      ao[j] = p;
    }
  }
  __syncthreads();
  { int i4 = tid * 4;                                // zero ctx accumulator
    ctx[i4] = 0.f; ctx[i4 + 1] = 0.f; ctx[i4 + 2] = 0.f; ctx[i4 + 3] = 0.f; }
  __syncthreads();

  // Phase 3: context = P(16x2048, f32 in LDS) @ V(2048x64, bf16 transposed).
  v8f cv[4]; { v8f zz = {}; cv[0]=zz; cv[1]=zz; cv[2]=zz; cv[3]=zz; }
  for (int kc = 0; kc < 8; ++kc) {
    int kb0 = wid * 256 + kc * 32;
    v16bf ap = frag_f32(sc + l15 * 2048 + kb0 + cb);
#pragma unroll
    for (int t = 0; t < 4; ++t) {
      const bf16* vp = vT + (size_t)(bh * DKV + t * 16 + l15) * SS + kb0 + cb;
      v16bf bv = frag_bf16(vp);
      cv[t] = wmma_bf16(ap, bv, cv[t]);
    }
  }
#pragma unroll
  for (int t = 0; t < 4; ++t)
#pragma unroll
    for (int r = 0; r < 8; ++r) {
      int m = r + hi;
      atomicAdd(&ctx[m * 64 + t * 16 + l15], cv[t][r]);   // ds_add_f32
    }
  __syncthreads();
  { int i4 = tid * 4;
    int m = i4 >> 6, col = i4 & 63;
    bf16* dst = ctxbuf + (size_t)(b * SS + m0 + m) * DD + h * DKV + col;
#pragma unroll
    for (int u = 0; u < 4; ++u) dst[u] = (bf16)ctx[m * 64 + col + u]; }
}

// ---------------- Kernel 3: ctx @ Wo + bo + residual, then LayerNorm ---------
__global__ void __launch_bounds__(256) out_kernel(
    const bf16* __restrict__ ctxbuf, const bf16* __restrict__ woT,
    const float* __restrict__ bo, const float* __restrict__ resid,
    const float* __restrict__ gamma, const float* __restrict__ beta,
    float* __restrict__ out) {
  extern __shared__ char smem[];
  float* xr = (float*)smem;                  // [16][1024] f32 strip
  int tid = threadIdx.x, wid = tid >> 5, lane = tid & 31;
  int rows0 = blockIdx.x * 16;
  int cb = (lane & 16) ? 8 : 0;
  int l15 = lane & 15;
  int hi = (lane & 16) ? 8 : 0;

  v8f acc[8]; { v8f zz = {};
#pragma unroll
    for (int t = 0; t < 8; ++t) acc[t] = zz; }

  const bf16* aBase = ctxbuf + (size_t)(rows0 + l15) * DD + cb;
  for (int kk = 0; kk < DD; kk += 32) {
    __builtin_prefetch(aBase + kk + 64, 0, 0);
    v16bf a = frag_bf16(aBase + kk);
#pragma unroll
    for (int t = 0; t < 8; ++t) {
      int n = wid * 128 + t * 16 + l15;
      v16bf bfr = frag_bf16(woT + (size_t)n * DD + kk + cb);
      acc[t] = wmma_bf16(a, bfr, acc[t]);
    }
  }
#pragma unroll
  for (int t = 0; t < 8; ++t) {
    int n = wid * 128 + t * 16 + l15;
    float bv = bo[n];
#pragma unroll
    for (int r = 0; r < 8; ++r) {
      int m = r + hi;
      xr[m * 1024 + n] = acc[t][r] + bv + resid[(size_t)(rows0 + m) * DD + n];
    }
  }
  __syncthreads();
#pragma unroll
  for (int e = 0; e < 2; ++e) {
    int r = wid * 2 + e;
    const float* row = xr + r * 1024;
    float s = 0.f, ss = 0.f;
    for (int j = lane; j < 1024; j += 32) { float v = row[j]; s += v; ss += v * v; }
    s = wave_sum(s); ss = wave_sum(ss);
    float mu = s * (1.0f / 1024.0f);
    float var = ss * (1.0f / 1024.0f) - mu * mu;
    float inv = rsqrtf(var + 1e-5f);
    float* op = out + (size_t)(rows0 + r) * DD;
    for (int j = lane; j < 1024; j += 32)
      op[j] = (row[j] - mu) * inv * gamma[j] + beta[j];
  }
}

extern "C" void kernel_launch(void* const* d_in, const int* in_sizes, int n_in,
                              void* d_out, int out_size, void* d_ws, size_t ws_size,
                              hipStream_t stream) {
  const float* Q  = (const float*)d_in[0];
  const float* K  = (const float*)d_in[1];
  const float* V  = (const float*)d_in[2];
  const unsigned char* mask = (const unsigned char*)d_in[3]; // jax bool = 1 byte
  const float* Wq = (const float*)d_in[4];
  const float* bq = (const float*)d_in[5];
  const float* Wk = (const float*)d_in[6];
  const float* bk = (const float*)d_in[7];
  const float* Wv = (const float*)d_in[8];
  const float* bv = (const float*)d_in[9];
  const float* Wo = (const float*)d_in[10];
  const float* bo = (const float*)d_in[11];
  const float* gamma = (const float*)d_in[12];
  const float* beta  = (const float*)d_in[13];

  // workspace: 5 x 8 MB bf16 buffers = 40 MB
  char* ws = (char*)d_ws;
  bf16* wT   = (bf16*)(ws);                      // [4][1024][1024] transposed weights
  bf16* qbuf = (bf16*)(ws + ((size_t)8  << 20)); // [4096][1024]
  bf16* kbuf = (bf16*)(ws + ((size_t)16 << 20)); // [4096][1024]
  bf16* vT   = (bf16*)(ws + ((size_t)24 << 20)); // [B*H*64][2048]
  bf16* ctxb = (bf16*)(ws + ((size_t)32 << 20)); // [4096][1024]

  float* out  = (float*)d_out;
  float* attn = out + (size_t)BB * SS * DD;      // out then attn, flat

  wt_kernel  <<<dim3(4096, 4), 256, 0, stream>>>(Wq, Wk, Wv, Wo, wT);
  proj_kernel<<<dim3(512, 3),  256, 0, stream>>>(Q, K, V, wT, bq, bk, bv,
                                                 qbuf, kbuf, vT);
  size_t sm2 = (size_t)16 * 2048 * 4 + 16 * 64 * 2 + 16 * 64 * 4;  // 137,216 B
  attn_kernel<<<dim3(4096), 256, sm2, stream>>>(qbuf, kbuf, vT, mask, ctxb, attn);
  size_t sm3 = (size_t)16 * 1024 * 4;                              // 65,536 B
  out_kernel <<<dim3(256), 256, sm3, stream>>>(ctxb, wT + (size_t)3 * DD * DD,
                                               bo, Q, gamma, beta, out);
}
